// JANET_88175678587046
// MI455X (gfx1250) — compile-verified, matrix operand order checked by reading, not run on hardware
//
#include <hip/hip_runtime.h>
#include <hip/hip_bf16.h>

// ---------------------------------------------------------------------------
// Types
// ---------------------------------------------------------------------------
typedef __attribute__((ext_vector_type(16))) __bf16        v16bf;
typedef __attribute__((ext_vector_type(8)))  float         v8f;
typedef __attribute__((ext_vector_type(8)))  unsigned int  v8u;
typedef __attribute__((ext_vector_type(4)))  unsigned int  v4u;
typedef __attribute__((ext_vector_type(4)))  unsigned int  u32x4;
typedef __attribute__((ext_vector_type(4)))  int           i32x4;
typedef __attribute__((ext_vector_type(8)))  int           i32x8;

#define RT   4096
#define RD   2048
#define RH   2048
#define NBLK 128      // persistent blocks in the recurrence kernel
#define SLICE 16      // H / NBLK output columns per block
#define KC   128      // GEMM K-chunk staged in LDS per TDM transfer

#if defined(__AMDGCN__) && __has_builtin(__builtin_amdgcn_tensor_load_to_lds)
#define HAVE_TDM 1
#endif

__device__ __forceinline__ unsigned short f2bf(float f) {
    unsigned u = __builtin_bit_cast(unsigned, f);
    unsigned r = u + 0x7FFFu + ((u >> 16) & 1u);   // round-to-nearest-even
    return (unsigned short)(r >> 16);
}

#ifdef HAVE_TDM
// ---------------------------------------------------------------------------
// Issue a TDM 2-D tile load (bf16 elements) into LDS.
// D# layout per CDNA5 ISA 08_async_tensor §8.3/8.4:
//   group0: count=1 | lds_addr | global_addr[56:0] | type=2
//   group1: data_size=1(2B), tensor_dim0/1, tile_dim0/1, tensor_dim0_stride
// 2-D tensor -> groups 2/3 unused (zero).
// ---------------------------------------------------------------------------
__device__ __forceinline__ void tdm_load_2d(unsigned lds_addr,
                                            const void* gptr,
                                            unsigned tile_d0, unsigned tile_d1,
                                            unsigned tensor_d0, unsigned tensor_d1,
                                            unsigned long long stride0_elems)
{
    unsigned long long ga = (unsigned long long)gptr;
    u32x4 g0;
    g0.x = 1u;                                                    // count=1, user mode
    g0.y = lds_addr;                                              // LDS byte address
    g0.z = (unsigned)ga;                                          // global_addr[31:0]
    g0.w = (unsigned)((ga >> 32) & 0x01FFFFFFull) | (2u << 30);   // addr[56:32] | type=2
    unsigned d0 = (1u << 16);                                     // data_size=1 -> 2 bytes
    unsigned d1 = (tensor_d0 & 0xFFFFu) << 16;                    // tensor_dim0[15:0]
    unsigned d2 = (tensor_d0 >> 16) | ((tensor_d1 & 0xFFFFu) << 16);
    unsigned d3 = (tensor_d1 >> 16) | (tile_d0 << 16);            // tile_dim0
    unsigned d4 = (tile_d1 & 0xFFFFu);                            // tile_dim1 (tile_dim2=0)
    unsigned d5 = (unsigned)(stride0_elems & 0xFFFFFFFFull);      // dim0 stride[31:0]
    unsigned d6 = (unsigned)((stride0_elems >> 32) & 0xFFFFull);  // dim0 stride[47:32]
    i32x8 g1 = { (int)d0, (int)d1, (int)d2, (int)d3,
                 (int)d4, (int)d5, (int)d6, 0 };
    i32x4 z4 = { 0, 0, 0, 0 };
#if __clang_major__ >= 23
    i32x8 z8 = { 0, 0, 0, 0, 0, 0, 0, 0 };
    __builtin_amdgcn_tensor_load_to_lds(g0, g1, z4, z4, z8, 0);
#else
    __builtin_amdgcn_tensor_load_to_lds(g0, g1, z4, z4, 0);
#endif
}

// Wait until TENSORcnt <= N (N must be an immediate for the builtin).
template <int N>
__device__ __forceinline__ void tdm_wait() {
#if __has_builtin(__builtin_amdgcn_s_wait_tensorcnt)
    __builtin_amdgcn_s_wait_tensorcnt(N);
#else
    asm volatile("s_wait_tensorcnt %0" :: "i"(N) : "memory");
#endif
}
#endif // HAVE_TDM

// ---------------------------------------------------------------------------
// fp32 -> bf16 conversion (plain, and transposing for GEMM B-operand)
// ---------------------------------------------------------------------------
__global__ void k_cvt_bf16(const float* __restrict__ in,
                           unsigned short* __restrict__ out, int n) {
    int i = blockIdx.x * blockDim.x + threadIdx.x;
    if (i < n) out[i] = f2bf(in[i]);
}

// in: [rowsIn, colsIn] row-major; out: [colsIn, rowsIn] row-major (transposed)
__global__ void k_cvt_bf16_T(const float* __restrict__ in,
                             unsigned short* __restrict__ out,
                             int rowsIn, int colsIn) {
    int i = blockIdx.x * blockDim.x + threadIdx.x;
    if (i < rowsIn * colsIn) {
        int n = i / rowsIn;      // column of input == row of output
        int r = i % rowsIn;
        out[i] = f2bf(in[(size_t)r * colsIn + n]);
    }
}

// ---------------------------------------------------------------------------
// GEMM: C[M,N] = A[M,K](bf16) * Bt[N,K](bf16, pre-transposed) + bias[N]
// Block tile 16(M) x 64(N), 4 waves. Double-buffered TDM pipeline:
// at chunk k, wave0 issues chunk k+1's two tensor_load_to_lds into the
// ping-pong buffer and waits TENSORcnt<=2 (in-order completion => chunk k's
// pair is done while k+1 streams), so WMMA overlaps the DMA.
// ---------------------------------------------------------------------------
__global__ __launch_bounds__(128) void k_gemm_bf16_wmma(
    const unsigned short* __restrict__ A,
    const unsigned short* __restrict__ Bt,
    const float* __restrict__ bias,
    float* __restrict__ C,
    int M, int N, int K)
{
    __shared__ unsigned short Atile[2][16 * KC];   // 2 x 4 KB
    __shared__ unsigned short Btile[2][64 * KC];   // 2 x 16 KB

    const int tid  = threadIdx.x;
    const int lane = tid & 31;
    const int wave = tid >> 5;
    const int l    = lane & 15;
    const int half = lane >> 4;
    const int m0   = blockIdx.x * 16;
    const int nblk = blockIdx.y * 64;
    const int n0   = nblk + wave * 16;

    const float bv = bias[n0 + l];
    v8f acc = { bv, bv, bv, bv, bv, bv, bv, bv };

    const int nch = K / KC;

#ifdef HAVE_TDM
    unsigned ldsA[2], ldsB[2];
    ldsA[0] = (unsigned)(unsigned long long)(void*)Atile[0];
    ldsA[1] = (unsigned)(unsigned long long)(void*)Atile[1];
    ldsB[0] = (unsigned)(unsigned long long)(void*)Btile[0];
    ldsB[1] = (unsigned)(unsigned long long)(void*)Btile[1];

    if (tid < 32) {   // prologue: start chunk 0
        tdm_load_2d(ldsA[0], A + (size_t)m0 * K, KC, 16,
                    (unsigned)K, (unsigned)M, (unsigned long long)K);
        tdm_load_2d(ldsB[0], Bt + (size_t)nblk * K, KC, 64,
                    (unsigned)K, (unsigned)N, (unsigned long long)K);
    }
#endif

    for (int ch = 0; ch < nch; ++ch) {
        const int cur = ch & 1;
        const int k0  = ch * KC;

#ifdef HAVE_TDM
        if (tid < 32) {
            if (ch + 1 < nch) {
                // stream next chunk into the other buffer while we compute
                tdm_load_2d(ldsA[cur ^ 1], A + (size_t)m0 * K + k0 + KC, KC, 16,
                            (unsigned)K, (unsigned)M, (unsigned long long)K);
                tdm_load_2d(ldsB[cur ^ 1], Bt + (size_t)nblk * K + k0 + KC, KC, 64,
                            (unsigned)K, (unsigned)N, (unsigned long long)K);
                tdm_wait<2>();   // chunk ch's pair complete; ch+1 still in flight
            } else {
                tdm_wait<0>();
            }
        }
#else
        // Fallback: cooperative global->LDS copy (16-byte chunks)
        for (int idx = tid; idx < 16 * (KC / 8); idx += 128) {
            int row = idx >> 3, d = idx & 7;
            *(v4u*)(&Atile[cur][row * KC + d * 8]) =
                *(const v4u*)(A + (size_t)(m0 + row) * K + k0 + d * 8);
        }
        for (int idx = tid; idx < 64 * (KC / 8); idx += 128) {
            int row = (idx >> 4), d = idx & 15;
            *(v4u*)(&Btile[cur][row * KC + d * 8]) =
                *(const v4u*)(Bt + (size_t)(nblk + row) * K + k0 + d * 8);
        }
#endif
        __syncthreads();   // tiles for chunk ch visible to all waves

        if (tid == 0 && ch + 2 < nch)
            __builtin_prefetch(A + (size_t)m0 * K + k0 + 2 * KC, 0, 3);  // global_prefetch_b8

        const unsigned short* Ar = &Atile[cur][l * KC + 8 * half];
        const unsigned short* Br = &Btile[cur][(wave * 16 + l) * KC + 8 * half];

#pragma unroll
        for (int kk = 0; kk < KC; kk += 32) {
            v4u alo = *(const v4u*)(Ar + kk);
            v4u ahi = *(const v4u*)(Ar + kk + 16);
            v4u blo = *(const v4u*)(Br + kk);
            v4u bhi = *(const v4u*)(Br + kk + 16);
            v8u au; au.lo = alo; au.hi = ahi;
            v8u bu; bu.lo = blo; bu.hi = bhi;
            acc = __builtin_amdgcn_wmma_f32_16x16x32_bf16(
                      false, __builtin_bit_cast(v16bf, au),
                      false, __builtin_bit_cast(v16bf, bu),
                      (short)0, acc, false, false);
        }
        __syncthreads();   // done reading buf[cur] before it is re-filled
    }

    float* Crow = C + (size_t)(m0 + 8 * half) * N + n0 + l;
#pragma unroll
    for (int r = 0; r < 8; ++r)
        Crow[(size_t)r * N] = acc[r];
}

// ---------------------------------------------------------------------------
// Persistent JANET recurrence. NBLK blocks; block b keeps the 16-column
// slices of U_f / U_h resident in LDS (2 x 64 KB, bf16) — possible due to
// CDNA5's 320 KB-per-WGP LDS. Grid sync on a monotonic atomic counter with
// s_sleep backoff.
// ---------------------------------------------------------------------------
__global__ __launch_bounds__(256) void k_janet_rec(
    const float* __restrict__ xf, const float* __restrict__ xh,
    const unsigned short* __restrict__ Uf, const unsigned short* __restrict__ Uh,
    const float* __restrict__ prelu,
    float* __restrict__ hs,          // [T,H] all hidden states
    float* __restrict__ hbuf,        // [2,H] double-buffered h broadcast (zeroed)
    unsigned int* __restrict__ syncc,
    int T)
{
    extern __shared__ unsigned int lds_u[];     // [RH*8] Uf dwords, [RH*8] Uh dwords
    unsigned int* Ufl = lds_u;
    unsigned int* Uhl = lds_u + RH * 8;
    __shared__ float redf[256], redh[256];

    const int tid = threadIdx.x;
    const int blk = blockIdx.x;

    const unsigned int* Ufd = (const unsigned int*)Uf;
    const unsigned int* Uhd = (const unsigned int*)Uh;
    for (int idx = tid; idx < RH * 8; idx += 256) {
        int row = idx >> 3, d = idx & 7;
        Ufl[idx] = Ufd[(size_t)row * (RH / 2) + blk * 8 + d];
        Uhl[idx] = Uhd[(size_t)row * (RH / 2) + blk * 8 + d];
    }
    __syncthreads();

    const float a   = prelu[0];
    const int c     = tid & 15;          // column within slice
    const int g     = tid >> 4;          // partial-sum group 0..15
    const int i0    = g * (RH / 16);     // 128 rows per group
    const int shift = (c & 1) * 16;
    const int dcol  = c >> 1;

    for (int t = 0; t < T; ++t) {
        const float* hp = hbuf + (size_t)(t & 1) * RH;
        float sf = 0.f, sh = 0.f;
#pragma unroll 4
        for (int i = i0; i < i0 + RH / 16; ++i) {
            float hv = hp[i];
            unsigned uf = Ufl[i * 8 + dcol];
            unsigned uh = Uhl[i * 8 + dcol];
            sf = fmaf(hv, __builtin_bit_cast(float, ((uf >> shift) & 0xFFFFu) << 16), sf);
            sh = fmaf(hv, __builtin_bit_cast(float, ((uh >> shift) & 0xFFFFu) << 16), sh);
        }
        redf[tid] = sf; redh[tid] = sh;
        __syncthreads();

        if (tid < 16) {
            float af = 0.f, ah = 0.f;
#pragma unroll
            for (int g2 = 0; g2 < 16; ++g2) {
                af += redf[g2 * 16 + tid];
                ah += redh[g2 * 16 + tid];
            }
            const int col = blk * SLICE + tid;
            const float hprev = hp[col];
            const float fg   = 1.f / (1.f + __expf(-(af + xf[(size_t)t * RH + col])));
            const float cand = tanhf(ah + xh[(size_t)t * RH + col]);
            float hn = fg * hprev + (1.f - fg) * cand;
            hn = (hn >= 0.f) ? hn : a * hn;               // PReLU
            hs[(size_t)t * RH + col]               = hn;
            hbuf[(size_t)((t + 1) & 1) * RH + col] = hn;
            __threadfence();
        }
        __syncthreads();

        if (tid == 0) {
            atomicAdd(syncc, 1u);
            const unsigned target = (unsigned)(t + 1) * NBLK;
            while (*(volatile unsigned int*)syncc < target)
                __builtin_amdgcn_s_sleep(2);
        }
        __syncthreads();
    }
}

// ---------------------------------------------------------------------------
// LayerNorm (unbiased std, torch semantics). Writes f32 and bf16 outputs.
// ---------------------------------------------------------------------------
__global__ __launch_bounds__(256) void k_layernorm(
    const float* __restrict__ x, const float* __restrict__ g,
    const float* __restrict__ b,
    float* __restrict__ y32, unsigned short* __restrict__ y16, int H)
{
    __shared__ float s1[256], s2[256];
    const int t = blockIdx.x, tid = threadIdx.x;
    const float* row = x + (size_t)t * H;
    float a0 = 0.f, a1 = 0.f;
    for (int j = tid; j < H; j += 256) { float v = row[j]; a0 += v; a1 += v * v; }
    s1[tid] = a0; s2[tid] = a1; __syncthreads();
    for (int s = 128; s > 0; s >>= 1) {
        if (tid < s) { s1[tid] += s1[tid + s]; s2[tid] += s2[tid + s]; }
        __syncthreads();
    }
    const float mean = s1[0] / (float)H;
    const float var  = (s2[0] - (float)H * mean * mean) / (float)(H - 1);
    const float inv  = 1.0f / (sqrtf(fmaxf(var, 0.f)) + 1e-6f);
    for (int j = tid; j < H; j += 256) {
        float v = (row[j] - mean) * inv * g[j] + b[j];
        y32[(size_t)t * H + j] = v;
        y16[(size_t)t * H + j] = f2bf(v);
    }
}

// ---------------------------------------------------------------------------
// Output head: logits = hn @ W_out + b_out, then log_softmax. One block/row.
// ---------------------------------------------------------------------------
__global__ __launch_bounds__(256) void k_head(
    const float* __restrict__ hn, const float* __restrict__ W,
    const float* __restrict__ b, float* __restrict__ out, int H)
{
    __shared__ float red[256];
    __shared__ float lg[16];
    const int t = blockIdx.x, tid = threadIdx.x;
    const float* row = hn + (size_t)t * H;
    float acc[10];
#pragma unroll
    for (int o = 0; o < 10; ++o) acc[o] = 0.f;
    for (int j = tid; j < H; j += 256) {
        float v = row[j];
#pragma unroll
        for (int o = 0; o < 10; ++o)
            acc[o] = fmaf(v, W[(size_t)j * 10 + o], acc[o]);
    }
    for (int o = 0; o < 10; ++o) {
        red[tid] = acc[o]; __syncthreads();
        for (int s = 128; s > 0; s >>= 1) {
            if (tid < s) red[tid] += red[tid + s];
            __syncthreads();
        }
        if (tid == 0) lg[o] = red[0] + b[o];
        __syncthreads();
    }
    if (tid == 0) {
        float m = -1e30f;
        for (int o = 0; o < 10; ++o) m = fmaxf(m, lg[o]);
        float s = 0.f;
        for (int o = 0; o < 10; ++o) s += __expf(lg[o] - m);
        const float ls = __logf(s);
        for (int o = 0; o < 10; ++o) out[(size_t)t * 10 + o] = lg[o] - m - ls;
    }
}

// ---------------------------------------------------------------------------
// Orchestration
// ---------------------------------------------------------------------------
extern "C" void kernel_launch(void* const* d_in, const int* in_sizes, int n_in,
                              void* d_out, int out_size, void* d_ws, size_t ws_size,
                              hipStream_t stream) {
    const float* x    = (const float*)d_in[0];
    const float* Wf0  = (const float*)d_in[1];
    const float* Uf0  = (const float*)d_in[2];
    const float* bf0  = (const float*)d_in[3];
    const float* Wh0  = (const float*)d_in[4];
    const float* Uh0  = (const float*)d_in[5];
    const float* bh0  = (const float*)d_in[6];
    const float* pr0  = (const float*)d_in[7];
    const float* ga0  = (const float*)d_in[8];
    const float* be0  = (const float*)d_in[9];
    const float* Wf1  = (const float*)d_in[10];
    const float* Uf1  = (const float*)d_in[11];
    const float* bf1  = (const float*)d_in[12];
    const float* Wh1  = (const float*)d_in[13];
    const float* Uh1  = (const float*)d_in[14];
    const float* bh1  = (const float*)d_in[15];
    const float* pr1  = (const float*)d_in[16];
    const float* ga1  = (const float*)d_in[17];
    const float* be1  = (const float*)d_in[18];
    const float* Wout = (const float*)d_in[19];
    const float* bout = (const float*)d_in[20];
    float* out = (float*)d_out;

    const int T = RT, D = RD, H = RH;

    char* ws = (char*)d_ws;
    size_t off = 0;
    unsigned short* xbf  = (unsigned short*)(ws + off); off += (size_t)T * D * 2; // x/hn bf16
    unsigned short* Wtf  = (unsigned short*)(ws + off); off += (size_t)H * D * 2; // W_f^T bf16
    unsigned short* Wth  = (unsigned short*)(ws + off); off += (size_t)H * D * 2; // W_h^T bf16
    unsigned short* Ufb  = (unsigned short*)(ws + off); off += (size_t)H * H * 2; // U_f bf16
    unsigned short* Uhb  = (unsigned short*)(ws + off); off += (size_t)H * H * 2; // U_h bf16
    float* xfb  = (float*)(ws + off); off += (size_t)T * H * 4;
    float* xhb  = (float*)(ws + off); off += (size_t)T * H * 4;
    float* hsb  = (float*)(ws + off); off += (size_t)T * H * 4;
    float* hn32 = (float*)(ws + off); off += (size_t)T * H * 4;
    float* hbuf = (float*)(ws + off); off += (size_t)2 * H * 4;
    unsigned int* syncc = (unsigned int*)(ws + off); off += 256;

    (void)hipFuncSetAttribute((const void*)k_janet_rec,
                              hipFuncAttributeMaxDynamicSharedMemorySize,
                              2 * RH * 8 * 4);

    const int cvtB = 256;
    dim3 gGemm(T / 16, H / 64);

    // ---- Layer 0 ----
    k_cvt_bf16  <<<(T * D + cvtB - 1) / cvtB, cvtB, 0, stream>>>(x,   xbf, T * D);
    k_cvt_bf16_T<<<(D * H + cvtB - 1) / cvtB, cvtB, 0, stream>>>(Wf0, Wtf, D, H);
    k_cvt_bf16_T<<<(D * H + cvtB - 1) / cvtB, cvtB, 0, stream>>>(Wh0, Wth, D, H);
    k_cvt_bf16  <<<(H * H + cvtB - 1) / cvtB, cvtB, 0, stream>>>(Uf0, Ufb, H * H);
    k_cvt_bf16  <<<(H * H + cvtB - 1) / cvtB, cvtB, 0, stream>>>(Uh0, Uhb, H * H);

    k_gemm_bf16_wmma<<<gGemm, 128, 0, stream>>>(xbf, Wtf, bf0, xfb, T, H, D);
    k_gemm_bf16_wmma<<<gGemm, 128, 0, stream>>>(xbf, Wth, bh0, xhb, T, H, D);

    (void)hipMemsetAsync(hbuf, 0, (size_t)2 * H * 4 + 256, stream);   // hbuf + sync counter
    k_janet_rec<<<NBLK, 256, 2 * RH * 8 * 4, stream>>>(xfb, xhb, Ufb, Uhb, pr0,
                                                       hsb, hbuf, syncc, T);
    k_layernorm<<<T, 256, 0, stream>>>(hsb, ga0, be0, hn32, xbf, H);

    // ---- Layer 1 (reuse weight buffers; xbf now holds normalized h in bf16) ----
    k_cvt_bf16_T<<<(H * H + cvtB - 1) / cvtB, cvtB, 0, stream>>>(Wf1, Wtf, H, H);
    k_cvt_bf16_T<<<(H * H + cvtB - 1) / cvtB, cvtB, 0, stream>>>(Wh1, Wth, H, H);
    k_cvt_bf16  <<<(H * H + cvtB - 1) / cvtB, cvtB, 0, stream>>>(Uf1, Ufb, H * H);
    k_cvt_bf16  <<<(H * H + cvtB - 1) / cvtB, cvtB, 0, stream>>>(Uh1, Uhb, H * H);

    k_gemm_bf16_wmma<<<gGemm, 128, 0, stream>>>(xbf, Wtf, bf1, xfb, T, H, H);
    k_gemm_bf16_wmma<<<gGemm, 128, 0, stream>>>(xbf, Wth, bh1, xhb, T, H, H);

    (void)hipMemsetAsync(hbuf, 0, (size_t)2 * H * 4 + 256, stream);
    k_janet_rec<<<NBLK, 256, 2 * RH * 8 * 4, stream>>>(xfb, xhb, Ufb, Uhb, pr1,
                                                       hsb, hbuf, syncc, T);
    k_layernorm<<<T, 256, 0, stream>>>(hsb, ga1, be1, hn32, xbf, H);

    // ---- Head ----
    k_head<<<T, 256, 0, stream>>>(hn32, Wout, bout, out, H);
}
